// PAM_Module_19481971655242
// MI455X (gfx1250) — matrix operand order repaired and checked by reading, not compiled, not run on hardware
//
#include <hip/hip_runtime.h>
#include <cstdint>
#include <cmath>

#define B_  4
#define C_  256
#define C4_ 64
#define N_  4096

typedef __attribute__((ext_vector_type(16))) __bf16         bfx16;
typedef __attribute__((ext_vector_type(16))) unsigned short usx16;
typedef __attribute__((ext_vector_type(8)))  unsigned short usx8;
typedef __attribute__((ext_vector_type(8)))  float          fx8;

// float -> bf16 (round-to-nearest-even)
__device__ __forceinline__ unsigned short f2bf(float f) {
  unsigned u = __builtin_bit_cast(unsigned, f);
  u += 0x7fffu + ((u >> 16) & 1u);
  return (unsigned short)(u >> 16);
}

// A-operand (16x32 bf16) tile load from a row-major [row][32] source.
// K order used here (same as ld_b16, not AMD's table order): lanes 0-15 -> rows, K contiguous {base..base+15};
// base = 0 for lanes 0-15, 16 for lanes 16-31.
__device__ __forceinline__ bfx16 ld_a16(const unsigned short* row, int sel8) {
  usx8 lo = *(const usx8*)(row + 2 * sel8);
  usx8 hi = *(const usx8*)(row + 2 * sel8 + 8);
  usx16 a;
#pragma unroll
  for (int e = 0; e < 8; ++e) { a[e] = lo[e]; a[8 + e] = hi[e]; }
  return __builtin_bit_cast(bfx16, a);
}

// B-operand (32x16 bf16): lane = column; lanes 0-15 hold K=0..15,
// lanes 16-31 hold K=16..31 -> 16 contiguous halves per lane.
__device__ __forceinline__ bfx16 ld_b16(const unsigned short* p) {
  usx16 b = *(const usx16*)p;
  return __builtin_bit_cast(bfx16, b);
}

// gfx1250 async DMA: global -> LDS, 16B per lane, tracked by ASYNCcnt.
__device__ __forceinline__ void async_b128(unsigned lds_off, const unsigned short* g) {
  asm volatile("global_load_async_to_lds_b128 %0, %1, off"
               :: "v"(lds_off), "v"(g) : "memory");
}
__device__ __forceinline__ void wait_async0() {
  asm volatile("s_wait_asynccnt 0x0" ::: "memory");
}

// ---------------------------------------------------------------- pack
__global__ void pack_kernel(const float* __restrict__ x,
                            const float* __restrict__ Wq,
                            const float* __restrict__ Wk,
                            const float* __restrict__ Wv,
                            unsigned short* __restrict__ XT,
                            unsigned short* __restrict__ WQ,
                            unsigned short* __restrict__ WK,
                            unsigned short* __restrict__ WV) {
  int idx = blockIdx.x * blockDim.x + threadIdx.x;
  const int xcnt = B_ * C_ * N_;
  if (idx < xcnt) {
    int b = idx / (C_ * N_);
    int r = idx % (C_ * N_);
    int c = r / N_;
    int n = r % N_;
    XT[((size_t)b * N_ + n) * C_ + c] = f2bf(x[idx]);   // x transposed -> [B][N][C]
  } else {
    int k = idx - xcnt;
    if (k < C4_ * C_)                     WQ[k]              = f2bf(Wq[k]);
    else if (k < 2 * C4_ * C_)            WK[k - C4_ * C_]   = f2bf(Wk[k - C4_ * C_]);
    else if (k < 2 * C4_ * C_ + C_ * C_)  WV[k - 2*C4_*C_]   = f2bf(Wv[k - 2*C4_*C_]);
  }
}

// ---------------------------------------------------------------- projections (WMMA)
__global__ __launch_bounds__(128) void proj_kernel(
    const unsigned short* __restrict__ XT,
    const unsigned short* __restrict__ WQ,
    const unsigned short* __restrict__ WK,
    const unsigned short* __restrict__ WV,
    const float* __restrict__ bq, const float* __restrict__ bk,
    const float* __restrict__ bv,
    unsigned short* __restrict__ QT, unsigned short* __restrict__ KT,
    unsigned short* __restrict__ V) {
  const int lane = threadIdx.x & 31;
  const int wave = blockIdx.x * 4 + (threadIdx.x >> 5);
  const int jobs_per_b = 1024 + 1024 + 4096;   // q-tiles + k-tiles + v-tiles
  const int b  = wave / jobs_per_b;
  int job      = wave % jobs_per_b;
  const unsigned short* W; const float* bias; int which;
  if (job < 1024)      { which = 0; W = WQ; bias = bq; }
  else if (job < 2048) { which = 1; W = WK; bias = bk; job -= 1024; }
  else                 { which = 2; W = WV; bias = bv; job -= 2048; }
  const int ot = job >> 8;       // output-channel tile
  const int nt = job & 255;      // position tile
  const int m15 = lane & 15, hi = lane >> 4;
  const int sel8 = hi * 8, sel16 = hi * 16;

  const unsigned short* wrow = W  + (size_t)(ot * 16 + m15) * C_;
  const unsigned short* xrow = XT + ((size_t)b * N_ + nt * 16 + m15) * C_;
  fx8 acc = {};
#pragma unroll
  for (int c0 = 0; c0 < C_; c0 += 32) {
    bfx16 a  = ld_a16(wrow + c0, sel8);
    bfx16 bb = ld_b16(xrow + c0 + sel16);
    acc = __builtin_amdgcn_wmma_f32_16x16x32_bf16(false, a, false, bb,
                                                  (short)0, acc, false, false);
  }
  const int n = nt * 16 + m15;
#pragma unroll
  for (int r = 0; r < 8; ++r) {
    int o = ot * 16 + r + 8 * hi;                 // D layout: M = r (+8 for hi half)
    unsigned short val = f2bf(acc[r] + bias[o]);
    if (which == 0)      QT[((size_t)b * N_ + n) * C4_ + o] = val;  // q^T [B][N][64]
    else if (which == 1) KT[((size_t)b * N_ + n) * C4_ + o] = val;  // k^T [B][N][64]
    else                 V [((size_t)b * C_ + o) * N_  + n] = val;  // v   [B][C][N]
  }
}

// ---------------------------------------------------------------- flash attention
// Block = 4 waves, each owning a 16-row i-tile. K/V j-tiles are shared by the
// whole block: double-buffered in LDS, filled with async global->LDS DMA.
__global__ __launch_bounds__(128) void attn_kernel(
    const unsigned short* __restrict__ QT,
    const unsigned short* __restrict__ KT,
    const unsigned short* __restrict__ V,
    const float* __restrict__ x, const float* __restrict__ gamma_p,
    float* __restrict__ out) {
  __shared__ unsigned short ktile[2][32 * 64];    // [jj][d]   4KB x2
  __shared__ unsigned short vtile[2][256 * 32];   // [c][jj]  16KB x2
  __shared__ unsigned short plds[4][16 * 32];     // per-wave P tile (16x32 bf16)

  const int tid  = threadIdx.x;
  const int lane = tid & 31;
  const int w    = tid >> 5;
  const int wave = blockIdx.x * 4 + w;
  const int b    = wave >> 8;                     // N/16 = 256 i-tiles per batch
  const int i0   = (wave & 255) * 16;
  const int m15 = lane & 15, hi = lane >> 4;
  const int sel8 = hi * 8, sel16 = hi * 16;
  const float gamma = gamma_p[0];

  // Q A-operand tiles for this 16-row i-tile (K = 64 channels -> 2 tiles)
  const unsigned short* qrow = QT + ((size_t)b * N_ + i0 + m15) * C4_;
  const bfx16 aq0 = ld_a16(qrow,      sel8);
  const bfx16 aq1 = ld_a16(qrow + 32, sel8);

  fx8   oacc[16] = {};            // O^T tile: 16 rows x 256 channels
  float mrow[8], lrow[8];
#pragma unroll
  for (int r = 0; r < 8; ++r) { mrow[r] = -INFINITY; lrow[r] = 0.f; }
  unsigned short* pt = &plds[w][0];

  const unsigned short* KTb = KT + (size_t)b * N_ * C4_;
  const unsigned short* Vb  = V  + (size_t)b * C_ * N_;

  // ---- async prefetch of one (K,V) j-tile into buffer t
  auto prefetch = [&](int t, int j0) {
    unsigned kbase = (unsigned)(size_t)&ktile[t][0];
    unsigned vbase = (unsigned)(size_t)&vtile[t][0];
    // K tile: 32 rows x 128B = 256 x 16B chunks, 2 per thread
#pragma unroll
    for (int q = 0; q < 2; ++q) {
      int chunk = tid * 2 + q;
      int row = chunk >> 3, part = chunk & 7;
      async_b128(kbase + (unsigned)(row * 64 + part * 8) * 2,
                 KTb + (size_t)(j0 + row) * C4_ + part * 8);
    }
    // V tile: 256 rows x 64B = 1024 x 16B chunks, 8 per thread
#pragma unroll
    for (int q = 0; q < 8; ++q) {
      int chunk = tid * 8 + q;
      int c = chunk >> 2, part = chunk & 3;
      async_b128(vbase + (unsigned)(c * 32 + part * 8) * 2,
                 Vb + (size_t)c * N_ + j0 + part * 8);
    }
  };

  prefetch(0, 0);

  for (int jt = 0; jt < N_ / 32; ++jt) {
    const int j0  = jt * 32;
    const int cur = jt & 1;
    wait_async0();          // my DMA for buffer `cur` has landed
    __syncthreads();        // everyone's DMA landed + everyone done reading cur^1
    if (jt + 1 < N_ / 32) prefetch(cur ^ 1, j0 + 32);

    const unsigned short* kt = &ktile[cur][0];
    fx8 s0 = {}, s1 = {};
    s0 = __builtin_amdgcn_wmma_f32_16x16x32_bf16(false, aq0, false,
            ld_b16(kt + m15 * 64 + sel16),            (short)0, s0, false, false);
    s0 = __builtin_amdgcn_wmma_f32_16x16x32_bf16(false, aq1, false,
            ld_b16(kt + m15 * 64 + 32 + sel16),       (short)0, s0, false, false);
    s1 = __builtin_amdgcn_wmma_f32_16x16x32_bf16(false, aq0, false,
            ld_b16(kt + (16 + m15) * 64 + sel16),     (short)0, s1, false, false);
    s1 = __builtin_amdgcn_wmma_f32_16x16x32_bf16(false, aq1, false,
            ld_b16(kt + (16 + m15) * 64 + 32 + sel16),(short)0, s1, false, false);

    // online softmax: row stats replicated across each 16-lane half
#pragma unroll
    for (int r = 0; r < 8; ++r) {
      float t = fmaxf(s0[r], s1[r]);
      t = fmaxf(t, __shfl_xor(t, 1)); t = fmaxf(t, __shfl_xor(t, 2));
      t = fmaxf(t, __shfl_xor(t, 4)); t = fmaxf(t, __shfl_xor(t, 8));
      float mnew = fmaxf(mrow[r], t);
      float al   = __expf(mrow[r] - mnew);
      float p0   = __expf(s0[r] - mnew);
      float p1   = __expf(s1[r] - mnew);
      float rs   = p0 + p1;
      rs += __shfl_xor(rs, 1); rs += __shfl_xor(rs, 2);
      rs += __shfl_xor(rs, 4); rs += __shfl_xor(rs, 8);
      lrow[r] = lrow[r] * al + rs;
      mrow[r] = mnew;
#pragma unroll
      for (int ct = 0; ct < 16; ++ct) oacc[ct][r] *= al;
      int prow = (r + 8 * hi) * 32;               // D layout -> row-major LDS
      pt[prow + m15]      = f2bf(p0);
      pt[prow + 16 + m15] = f2bf(p1);
    }
    asm volatile("s_wait_dscnt 0x0" ::: "memory");  // cross-lane RAW through LDS
    const bfx16 ap = ld_a16(pt + m15 * 32, sel8);   // P as A operand (16x32)

    const unsigned short* vt = &vtile[cur][0];
#pragma unroll
    for (int ct = 0; ct < 16; ++ct) {
      oacc[ct] = __builtin_amdgcn_wmma_f32_16x16x32_bf16(false, ap, false,
                    ld_b16(vt + (ct * 16 + m15) * 32 + sel16),
                    (short)0, oacc[ct], false, false);
    }
    asm volatile("" ::: "memory");
  }

#pragma unroll
  for (int ct = 0; ct < 16; ++ct) {
#pragma unroll
    for (int r = 0; r < 8; ++r) {
      int c = ct * 16 + m15;
      int i = i0 + r + 8 * hi;
      size_t idx = ((size_t)b * C_ + c) * N_ + i;
      out[idx] = gamma * oacc[ct][r] / lrow[r] + x[idx];
    }
  }
}

// ---------------------------------------------------------------- launcher
extern "C" void kernel_launch(void* const* d_in, const int* in_sizes, int n_in,
                              void* d_out, int out_size, void* d_ws, size_t ws_size,
                              hipStream_t stream) {
  const float* x     = (const float*)d_in[0];
  const float* Wq    = (const float*)d_in[1];
  const float* bq    = (const float*)d_in[2];
  const float* Wk    = (const float*)d_in[3];
  const float* bk    = (const float*)d_in[4];
  const float* Wv    = (const float*)d_in[5];
  const float* bv    = (const float*)d_in[6];
  const float* gamma = (const float*)d_in[7];
  float* out = (float*)d_out;

  unsigned short* ws = (unsigned short*)d_ws;          // ~21 MB bf16 staging (fits L2)
  unsigned short* XT = ws;                             // [B][N][C]
  unsigned short* WQ = XT + (size_t)B_ * N_ * C_;
  unsigned short* WK = WQ + (size_t)C4_ * C_;
  unsigned short* WV = WK + (size_t)C4_ * C_;
  unsigned short* QT = WV + (size_t)C_ * C_;           // [B][N][64]
  unsigned short* KT = QT + (size_t)B_ * N_ * C4_;     // [B][N][64]
  unsigned short* V  = KT + (size_t)B_ * N_ * C4_;     // [B][C][N]

  int total1 = B_ * C_ * N_ + 2 * C4_ * C_ + C_ * C_;
  pack_kernel<<<(total1 + 255) / 256, 256, 0, stream>>>(x, Wq, Wk, Wv, XT, WQ, WK, WV);

  int waves2 = B_ * (1024 + 1024 + 4096);              // full waves only (EXEC all-1s)
  proj_kernel<<<waves2 / 4, 128, 0, stream>>>(XT, WQ, WK, WV, bq, bk, bv, QT, KT, V);

  int waves3 = B_ * (N_ / 16);
  attn_kernel<<<waves3 / 4, 128, 0, stream>>>(QT, KT, V, x, gamma, out);
}
